// SlowMemory_71098888618417
// MI455X (gfx1250) — matrix operand order repaired and checked by reading, not compiled, not run on hardware
//
#include <hip/hip_runtime.h>
#include <hip/hip_bf16.h>
#include <stdint.h>

// ---------------- problem constants ----------------
#define D_MODEL   1024
#define N_SLOTS   8
#define N_HEADS   16
#define HEAD_DIM  64
#define BATCH     64
#define SEQ       512
#define NGROUP    4          // batch groups of 16
#define WG_PER_GROUP 16      // one WG per head / per 64-col N slice

typedef __attribute__((ext_vector_type(16))) __bf16   v16bf;
typedef __attribute__((ext_vector_type(8)))  float    v8f;
typedef __attribute__((ext_vector_type(4)))  unsigned uint32x4;

union Frag32 { v16bf v; uint32x4 q[2]; };

// ---------------- fp32 <-> bf16 helpers ----------------
__device__ __forceinline__ unsigned short f2bf(float f) {
  union { float f; unsigned u; } c; c.f = f;
  unsigned u = c.u;
  return (unsigned short)((u + 0x7FFFu + ((u >> 16) & 1u)) >> 16);  // RNE
}
__device__ __forceinline__ float bf2f(unsigned short h) {
  union { unsigned u; float f; } c; c.u = ((unsigned)h) << 16;
  return c.f;
}

// A fragment (16x32 bf16): lane<16 holds row M=lane, K {k0..k0+7, k0+16..k0+23};
// lane>=16 holds row M=lane-16, K {k0+8..k0+15, k0+24..k0+31}. Two aligned 16B loads.
__device__ __forceinline__ v16bf load_a_frag(const unsigned short* __restrict__ row,
                                             int k0, int hi) {
  Frag32 f;
  const unsigned short* p = row + k0 + (hi << 3);
  f.q[0] = *(const uint32x4*)(p);
  f.q[1] = *(const uint32x4*)(p + 16);
  return f.v;
}
// B fragment: pre-swizzled tile of 512 halves; lane grabs 16 contiguous halves.
__device__ __forceinline__ v16bf load_b_frag(const unsigned short* __restrict__ tile,
                                             int lane) {
  Frag32 f;
  const uint32x4* p = (const uint32x4*)(tile + lane * 16);
  f.q[0] = p[0];
  f.q[1] = p[1];
  return f.v;
}

// release/acquire barrier over the 16 WGs of one batch group
__device__ __forceinline__ void group_barrier(unsigned* ctr, unsigned target) {
  __syncthreads();
  if (threadIdx.x == 0) {
    __threadfence();  // release prior global stores to agent scope
    __hip_atomic_fetch_add(ctr, 1u, __ATOMIC_RELEASE, __HIP_MEMORY_SCOPE_AGENT);
    while (__hip_atomic_load(ctr, __ATOMIC_ACQUIRE, __HIP_MEMORY_SCOPE_AGENT) < target) {
      __builtin_amdgcn_s_sleep(2);
    }
  }
  __syncthreads();
}

// ---------------- prep kernels ----------------
__global__ void cvt_bf16(const float* __restrict__ src, unsigned short* __restrict__ dst,
                         size_t n) {
  size_t i = (size_t)blockIdx.x * blockDim.x + threadIdx.x;
  if (i < n) dst[i] = f2bf(src[i]);
}

// W: [N][K] row-major fp32  ->  Wt: [K/32][N/16][512] bf16, WMMA-B fragment order.
__global__ void tile_weight(const float* __restrict__ W, unsigned short* __restrict__ Wt,
                            int N, int K) {
  size_t idx = (size_t)blockIdx.x * blockDim.x + threadIdx.x;
  size_t total = (size_t)N * K;
  if (idx >= total) return;
  int ntiles = N >> 4;
  size_t tileid = idx >> 9;
  int within = (int)(idx & 511);
  int lane = within >> 4, j = within & 15;
  int kt = (int)(tileid / ntiles), nt = (int)(tileid % ntiles);
  int k = kt * 32 + ((lane >> 4) << 3) + (j & 7) + ((j >> 3) << 4);   // lane>=16 -> local K 8..15, 24..31
  int n = nt * 16 + (lane & 15);
  Wt[idx] = f2bf(W[(size_t)n * K + k]);
}

__global__ void init_state(const float* __restrict__ memory_slots,
                           float* __restrict__ memf, unsigned short* __restrict__ membf,
                           unsigned* __restrict__ barriers) {
  size_t i = (size_t)blockIdx.x * blockDim.x + threadIdx.x;
  size_t total = (size_t)BATCH * N_SLOTS * D_MODEL;
  if (i < total) {
    size_t rem = i % ((size_t)N_SLOTS * D_MODEL);
    float v = memory_slots[rem];
    memf[i]  = v;
    membf[i] = f2bf(v);
  }
  if (i < 64) barriers[i] = 0u;
}

// ---------------- main persistent scan kernel ----------------
__global__ void __launch_bounds__(256)
slowmem_main(const unsigned short* __restrict__ xbf,
             const unsigned short* __restrict__ wqT,
             const unsigned short* __restrict__ wkT,
             const unsigned short* __restrict__ wvT,
             const unsigned short* __restrict__ woT,   // attn_out_w
             const unsigned short* __restrict__ wwT,   // write_w
             const unsigned short* __restrict__ wgT,   // gate_w  (K=2048)
             const unsigned short* __restrict__ woutT, // out_w   (K=2048)
             float* __restrict__ memf,
             unsigned short* __restrict__ membf,
             unsigned short* __restrict__ read_raw,    // [64][1024] bf16
             unsigned short* __restrict__ read_proj,   // [64][1024] bf16
             float* __restrict__ wwpart,               // [64 WG][128]
             unsigned* __restrict__ barriers,
             const float* __restrict__ in_proj_b,
             const float* __restrict__ attn_out_b,
             const float* __restrict__ gate_b,
             const float* __restrict__ write_b,
             const float* __restrict__ out_b,
             float* __restrict__ enh_out)              // d_out[B][T][D] (pre-LN)
{
  __shared__ float          lds_q[16 * 64];
  __shared__ float          lds_wc[16 * 64];
  __shared__ unsigned short lds_k[128 * 64];
  __shared__ unsigned short lds_v[128 * 64];
  __shared__ float          lds_sc[16 * 8];
  __shared__ float          lds_attn[16 * 8];
  __shared__ float          lds_out[16 * 64];
  __shared__ float          lds_gate[16 * 64];
  __shared__ float          lds_ww[16 * 8];

  const int g     = blockIdx.x >> 4;     // batch group
  const int slice = blockIdx.x & 15;     // N-slice / head id
  const int h     = slice;
  const int n0    = slice * 64;
  const int tid   = threadIdx.x;
  const int lane  = tid & 31;
  const int wave  = tid >> 5;
  const int r15   = lane & 15;
  const int hi    = lane >> 4;

  unsigned* ctr = barriers + g * 16;     // 64B stride per group counter
  unsigned bar_target = 0;

  const float* bq = in_proj_b;
  const float* bk = in_proj_b + D_MODEL;
  const float* bv = in_proj_b + 2 * D_MODEL;

  for (int t = 0; t < SEQ; ++t) {
    const unsigned short* chunk_row =
        xbf + ((size_t)(g * 16 + r15) * SEQ + t) * D_MODEL;

    // ===== Phase 1a: k (waves 0-3) and v (waves 4-7), 8-way M register blocking =====
    // Double-buffered: all 18 fragment loads for K-step kt+1 are issued before the
    // 8 dependency-free WMMAs of K-step kt, so load latency overlaps the matrix pipe.
    {
      const int isV   = wave >> 2;             // 0: k, 1: v
      const int ntloc = wave & 3;
      const int ntglob = h * 4 + ntloc;
      const unsigned short* Bt = isV ? wvT : wkT;
      const unsigned short* Arow0 = membf + (size_t)(g * 128 + r15) * D_MODEL;
      v8f acc[8];
      v8f z = {};
#pragma unroll
      for (int mt = 0; mt < 8; ++mt) acc[mt] = z;

      v16bf bcur = load_b_frag(Bt + (size_t)ntglob * 512, lane);
      v16bf acur[8];
#pragma unroll
      for (int mt = 0; mt < 8; ++mt)
        acur[mt] = load_a_frag(Arow0 + (size_t)(mt * 16) * D_MODEL, 0, hi);

      for (int kt = 0; kt < 31; ++kt) {
        if (kt + 4 < 32)
          __builtin_prefetch(Bt + ((size_t)((kt + 4) * 64 + ntglob)) * 512, 0, 1);
        v16bf bnx = load_b_frag(Bt + ((size_t)((kt + 1) * 64 + ntglob)) * 512, lane);
        v16bf anx[8];
#pragma unroll
        for (int mt = 0; mt < 8; ++mt)
          anx[mt] = load_a_frag(Arow0 + (size_t)(mt * 16) * D_MODEL, (kt + 1) * 32, hi);
#pragma unroll
        for (int mt = 0; mt < 8; ++mt)
          acc[mt] = __builtin_amdgcn_wmma_f32_16x16x32_bf16(false, acur[mt], false, bcur,
                                                            (short)0, acc[mt],
                                                            false, false);
        bcur = bnx;
#pragma unroll
        for (int mt = 0; mt < 8; ++mt) acur[mt] = anx[mt];
      }
#pragma unroll
      for (int mt = 0; mt < 8; ++mt)
        acc[mt] = __builtin_amdgcn_wmma_f32_16x16x32_bf16(false, acur[mt], false, bcur,
                                                          (short)0, acc[mt],
                                                          false, false);

      const int col = ntloc * 16 + r15;
      const float bias = (isV ? bv : bk)[h * 64 + col];
      unsigned short* dst = isV ? lds_v : lds_k;
#pragma unroll
      for (int mt = 0; mt < 8; ++mt)
#pragma unroll
        for (int rr = 0; rr < 8; ++rr)
          dst[(mt * 16 + rr + 8 * hi) * 64 + col] = f2bf(acc[mt][rr] + bias);
    }
    // ===== Phase 1b: q (waves 0-3) and wc (waves 4-7), double-buffered =====
    {
      const int isWc  = wave >> 2;             // 0: q, 1: wc
      const int ntloc = wave & 3;
      const int ntglob = slice * 4 + ntloc;    // h == slice
      const unsigned short* Bt = isWc ? wwT : wqT;
      v8f c = {};
      v16bf a = load_a_frag(chunk_row, 0, hi);
      v16bf b = load_b_frag(Bt + (size_t)ntglob * 512, lane);
      for (int kt = 0; kt < 31; ++kt) {
        v16bf an = load_a_frag(chunk_row, (kt + 1) * 32, hi);
        v16bf bn = load_b_frag(Bt + ((size_t)((kt + 1) * 64 + ntglob)) * 512, lane);
        c = __builtin_amdgcn_wmma_f32_16x16x32_bf16(false, a, false, b, (short)0, c,
                                                    false, false);
        a = an; b = bn;
      }
      c = __builtin_amdgcn_wmma_f32_16x16x32_bf16(false, a, false, b, (short)0, c,
                                                  false, false);
      const int col = ntloc * 16 + r15;
      if (!isWc) {
        float bias = bq[h * 64 + col];
#pragma unroll
        for (int rr = 0; rr < 8; ++rr) lds_q[(rr + 8 * hi) * 64 + col] = c[rr] + bias;
      } else {
        float bias = write_b[n0 + col];
#pragma unroll
        for (int rr = 0; rr < 8; ++rr) lds_wc[(rr + 8 * hi) * 64 + col] = c[rr] + bias;
      }
    }
    __syncthreads();

    // scores[b][s] = q_b . k_{b,s} / sqrt(dh)
    if (tid < 128) {
      int b = tid >> 3, s = tid & 7;
      const unsigned short* kr = lds_k + (b * 8 + s) * 64;
      const float* qr = lds_q + b * 64;
      float acc = 0.f;
      for (int d = 0; d < 64; ++d) acc += qr[d] * bf2f(kr[d]);
      lds_sc[tid] = acc * 0.125f;
    }
    __syncthreads();
    if (tid < 16) {          // softmax over S=8
      int b = tid;
      float mx = lds_sc[b * 8];
      for (int s = 1; s < 8; ++s) mx = fmaxf(mx, lds_sc[b * 8 + s]);
      float e[8], sum = 0.f;
      for (int s = 0; s < 8; ++s) { e[s] = __expf(lds_sc[b * 8 + s] - mx); sum += e[s]; }
      float inv = 1.f / sum;
      for (int s = 0; s < 8; ++s) lds_attn[b * 8 + s] = e[s] * inv;
    }
    __syncthreads();
    // read_raw slice for this head
    for (int i = tid; i < 1024; i += 256) {
      int b = i >> 6, d2 = i & 63;
      float acc = 0.f;
      for (int s = 0; s < 8; ++s)
        acc += lds_attn[b * 8 + s] * bf2f(lds_v[(b * 8 + s) * 64 + d2]);
      read_raw[(size_t)(g * 16 + b) * D_MODEL + h * 64 + d2] = f2bf(acc);
    }
    // partial write-score: wc_slice . mem_slice (full D reduced across WGs in phase 3)
    if (tid < 128) {
      int b = tid >> 3, s = tid & 7;
      const float* mrow = memf + ((size_t)((g * 16 + b) * 8 + s)) * D_MODEL + n0;
      const float* wcr  = lds_wc + b * 64;
      float acc = 0.f;
      for (int cc = 0; cc < 64; ++cc) acc += wcr[cc] * mrow[cc];
      wwpart[((size_t)(g * 16 + slice)) * 128 + tid] = acc * 0.03125f;  // 1/sqrt(1024)
    }
    bar_target += WG_PER_GROUP;
    group_barrier(ctr, bar_target);   // A: read_raw + wwpart visible

    // ===== Phase 2: read_proj = read_raw @ attn_out_w^T (this slice) =====
    if (wave < 4) {
      int ntloc = wave;
      int ntglob = slice * 4 + ntloc;
      const unsigned short* Arow = read_raw + (size_t)(g * 16 + r15) * D_MODEL;
      v8f c = {};
      v16bf a = load_a_frag(Arow, 0, hi);
      v16bf b = load_b_frag(woT + (size_t)ntglob * 512, lane);
      for (int kt = 0; kt < 31; ++kt) {
        v16bf an = load_a_frag(Arow, (kt + 1) * 32, hi);
        v16bf bn = load_b_frag(woT + ((size_t)((kt + 1) * 64 + ntglob)) * 512, lane);
        c = __builtin_amdgcn_wmma_f32_16x16x32_bf16(false, a, false, b, (short)0, c,
                                                    false, false);
        a = an; b = bn;
      }
      c = __builtin_amdgcn_wmma_f32_16x16x32_bf16(false, a, false, b, (short)0, c,
                                                  false, false);
      int col = ntloc * 16 + r15;
      float bias = attn_out_b[n0 + col];
#pragma unroll
      for (int rr = 0; rr < 8; ++rr) {
        int row = rr + 8 * hi;
        read_proj[(size_t)(g * 16 + row) * D_MODEL + n0 + col] = f2bf(c[rr] + bias);
      }
    }
    bar_target += WG_PER_GROUP;
    group_barrier(ctr, bar_target);   // B: read_proj visible

    // ===== Phase 3: out & gate (K=2048 over [chunk, read_proj]), mem update =====
    {
      int which = wave >> 2;              // 0=out, 1=gate
      int ntloc = wave & 3;
      int ntglob = slice * 4 + ntloc;
      const unsigned short* Bt = which ? wgT : woutT;
      const unsigned short* ArowR = read_proj + (size_t)(g * 16 + r15) * D_MODEL;
      auto a_at = [&](int kt) -> v16bf {
        return (kt < 32) ? load_a_frag(chunk_row, kt * 32, hi)
                         : load_a_frag(ArowR, (kt - 32) * 32, hi);
      };
      v8f c = {};
      v16bf a = a_at(0);
      v16bf b = load_b_frag(Bt + (size_t)ntglob * 512, lane);
      for (int kt = 0; kt < 63; ++kt) {
        v16bf an = a_at(kt + 1);
        v16bf bn = load_b_frag(Bt + ((size_t)((kt + 1) * 64 + ntglob)) * 512, lane);
        c = __builtin_amdgcn_wmma_f32_16x16x32_bf16(false, a, false, b, (short)0, c,
                                                    false, false);
        a = an; b = bn;
      }
      c = __builtin_amdgcn_wmma_f32_16x16x32_bf16(false, a, false, b, (short)0, c,
                                                  false, false);
      int col = ntloc * 16 + r15;
      float* dst = which ? lds_gate : lds_out;
#pragma unroll
      for (int rr = 0; rr < 8; ++rr) dst[(rr + 8 * hi) * 64 + col] = c[rr];
    }
    __syncthreads();
    // write-weight softmax: sum the 16 per-slice partials deterministically
    if (tid < 16) {
      int b = tid;
      float sc[8];
      for (int s = 0; s < 8; ++s) {
        float a = 0.f;
        for (int sl = 0; sl < 16; ++sl)
          a += wwpart[((size_t)(g * 16 + sl)) * 128 + b * 8 + s];
        sc[s] = a;
      }
      float mx = sc[0];
      for (int s = 1; s < 8; ++s) mx = fmaxf(mx, sc[s]);
      float sum = 0.f;
      for (int s = 0; s < 8; ++s) { sc[s] = __expf(sc[s] - mx); sum += sc[s]; }
      float inv = 1.f / sum;
      for (int s = 0; s < 8; ++s) lds_ww[b * 8 + s] = sc[s] * inv;
    }
    for (int i = tid; i < 1024; i += 256) {
      int b = i >> 6, cc = i & 63;
      float e = lds_out[i] + out_b[n0 + cc];
      enh_out[((size_t)(g * 16 + b) * SEQ + t) * D_MODEL + n0 + cc] = e;
      float gv = 1.f / (1.f + __expf(-(lds_gate[i] + gate_b[n0 + cc])));
      lds_gate[i] = gv * lds_wc[i];     // (gate * wc) slice
    }
    __syncthreads();
    for (int i = tid; i < 8192; i += 256) {
      int b = i >> 9, s = (i >> 6) & 7, cc = i & 63;
      float delta = lds_ww[b * 8 + s] * lds_gate[b * 64 + cc];
      size_t off = ((size_t)((g * 16 + b) * 8 + s)) * D_MODEL + n0 + cc;
      float nv = memf[off] + delta;
      memf[off]  = nv;
      membf[off] = f2bf(nv);
    }
    bar_target += WG_PER_GROUP;
    group_barrier(ctr, bar_target);   // C: mem updates visible for next step
  }
}

// ---------------- epilogue kernels ----------------
__global__ void __launch_bounds__(256)
ln_residual(const float* __restrict__ x, float* __restrict__ out,
            const float* __restrict__ gamma, const float* __restrict__ beta) {
  __shared__ float s1[256], s2[256];
  size_t row = blockIdx.x;
  const float* xr = x + row * D_MODEL;
  float* er = out + row * D_MODEL;
  float vals[4], sum = 0.f, sq = 0.f;
  for (int i = 0; i < 4; ++i) {
    int c = threadIdx.x + i * 256;
    float v = er[c] + xr[c];
    vals[i] = v; sum += v; sq += v * v;
  }
  s1[threadIdx.x] = sum; s2[threadIdx.x] = sq;
  __syncthreads();
  for (int off = 128; off > 0; off >>= 1) {
    if ((int)threadIdx.x < off) {
      s1[threadIdx.x] += s1[threadIdx.x + off];
      s2[threadIdx.x] += s2[threadIdx.x + off];
    }
    __syncthreads();
  }
  float mu  = s1[0] * (1.f / 1024.f);
  float var = s2[0] * (1.f / 1024.f) - mu * mu;
  float rs  = rsqrtf(var + 1e-5f);
  for (int i = 0; i < 4; ++i) {
    int c = threadIdx.x + i * 256;
    er[c] = (vals[i] - mu) * rs * gamma[c] + beta[c];
  }
}

__global__ void copy_mem_out(const float* __restrict__ memf, float* __restrict__ dst) {
  size_t i = (size_t)blockIdx.x * blockDim.x + threadIdx.x;
  if (i < (size_t)BATCH * N_SLOTS * D_MODEL) dst[i] = memf[i];
}

// ---------------- host driver ----------------
extern "C" void kernel_launch(void* const* d_in, const int* in_sizes, int n_in,
                              void* d_out, int out_size, void* d_ws, size_t ws_size,
                              hipStream_t stream) {
  (void)in_sizes; (void)n_in; (void)out_size; (void)ws_size;
  const float* x            = (const float*)d_in[0];
  const float* memory_slots = (const float*)d_in[1];
  const float* in_proj_w    = (const float*)d_in[2];
  const float* in_proj_b    = (const float*)d_in[3];
  const float* attn_out_w   = (const float*)d_in[4];
  const float* attn_out_b   = (const float*)d_in[5];
  const float* gate_w       = (const float*)d_in[6];
  const float* gate_b       = (const float*)d_in[7];
  const float* write_w      = (const float*)d_in[8];
  const float* write_b      = (const float*)d_in[9];
  const float* out_w        = (const float*)d_in[10];
  const float* out_b        = (const float*)d_in[11];
  const float* ln_g         = (const float*)d_in[12];
  const float* ln_b         = (const float*)d_in[13];
  float* out = (float*)d_out;

  char* ws = (char*)d_ws;
  size_t o = 0;
  auto take = [&](size_t bytes) -> char* {
    char* p = ws + o;
    o += bytes;
    o = (o + 255) & ~(size_t)255;
    return p;
  };
  unsigned short* xbf   = (unsigned short*)take((size_t)BATCH * SEQ * D_MODEL * 2);
  unsigned short* wqT   = (unsigned short*)take((size_t)D_MODEL * D_MODEL * 2);
  unsigned short* wkT   = (unsigned short*)take((size_t)D_MODEL * D_MODEL * 2);
  unsigned short* wvT   = (unsigned short*)take((size_t)D_MODEL * D_MODEL * 2);
  unsigned short* woT   = (unsigned short*)take((size_t)D_MODEL * D_MODEL * 2);
  unsigned short* wwT   = (unsigned short*)take((size_t)D_MODEL * D_MODEL * 2);
  unsigned short* wgT   = (unsigned short*)take((size_t)D_MODEL * 2 * D_MODEL * 2);
  unsigned short* woutT = (unsigned short*)take((size_t)D_MODEL * 2 * D_MODEL * 2);
  float*          memf  = (float*)take((size_t)BATCH * N_SLOTS * D_MODEL * 4);
  unsigned short* membf = (unsigned short*)take((size_t)BATCH * N_SLOTS * D_MODEL * 2);
  unsigned short* rraw  = (unsigned short*)take((size_t)BATCH * D_MODEL * 2);
  unsigned short* rproj = (unsigned short*)take((size_t)BATCH * D_MODEL * 2);
  float*          wwp   = (float*)take((size_t)64 * 128 * 4);
  unsigned*       bars  = (unsigned*)take(64 * sizeof(unsigned));

  size_t nx  = (size_t)BATCH * SEQ * D_MODEL;
  size_t ndd = (size_t)D_MODEL * D_MODEL;
  size_t n2d = (size_t)D_MODEL * 2 * D_MODEL;

  cvt_bf16<<<(unsigned)((nx + 255) / 256), 256, 0, stream>>>(x, xbf, nx);
  tile_weight<<<(unsigned)((ndd + 255) / 256), 256, 0, stream>>>(in_proj_w,           wqT, D_MODEL, D_MODEL);
  tile_weight<<<(unsigned)((ndd + 255) / 256), 256, 0, stream>>>(in_proj_w + ndd,     wkT, D_MODEL, D_MODEL);
  tile_weight<<<(unsigned)((ndd + 255) / 256), 256, 0, stream>>>(in_proj_w + 2 * ndd, wvT, D_MODEL, D_MODEL);
  tile_weight<<<(unsigned)((ndd + 255) / 256), 256, 0, stream>>>(attn_out_w,          woT, D_MODEL, D_MODEL);
  tile_weight<<<(unsigned)((ndd + 255) / 256), 256, 0, stream>>>(write_w,             wwT, D_MODEL, D_MODEL);
  tile_weight<<<(unsigned)((n2d + 255) / 256), 256, 0, stream>>>(gate_w,              wgT, D_MODEL, 2 * D_MODEL);
  tile_weight<<<(unsigned)((n2d + 255) / 256), 256, 0, stream>>>(out_w,               woutT, D_MODEL, 2 * D_MODEL);

  size_t nmem = (size_t)BATCH * N_SLOTS * D_MODEL;
  init_state<<<(unsigned)((nmem + 255) / 256), 256, 0, stream>>>(memory_slots, memf, membf, bars);

  slowmem_main<<<NGROUP * WG_PER_GROUP, 256, 0, stream>>>(
      xbf, wqT, wkT, wvT, woT, wwT, wgT, woutT,
      memf, membf, rraw, rproj, wwp, bars,
      in_proj_b, attn_out_b, gate_b, write_b, out_b, out);

  ln_residual<<<BATCH * SEQ, 256, 0, stream>>>(x, out, ln_g, ln_b);
  copy_mem_out<<<(unsigned)((nmem + 255) / 256), 256, 0, stream>>>(
      memf, out + (size_t)BATCH * SEQ * D_MODEL);
}
